// MultiHeadAttention_12979391168608
// MI455X (gfx1250) — matrix operand order repaired and checked by reading, not compiled, not run on hardware
//
#include <hip/hip_runtime.h>

// Problem constants (from reference): B,C,M,H,K,V = 2,2048,1024,16,64,64
#define B_ 2
#define C_ 2048
#define M_ 1024
#define H_ 16
#define V_ 64

typedef __attribute__((ext_vector_type(16))) __bf16 v16bf;
typedef __attribute__((ext_vector_type(8)))  float  v8f;
typedef __attribute__((ext_vector_type(4)))  unsigned int u32x4;

union FragB { v16bf v; u32x4 q[2]; __bf16 e[16]; };
union AccU  { v8f v; float f[8]; };

__device__ __forceinline__ v8f wmma_bf16(v16bf a, v16bf b, v8f c) {
  // D = A(16x32 bf16) x B(32x16 bf16) + C(16x16 f32)
  return __builtin_amdgcn_wmma_f32_16x16x32_bf16(false, a, false, b, (short)0, c,
                                                 false, false);
}

// --- CDNA5 async global->LDS copy (16B per lane), tracked by ASYNCcnt -------
__device__ __forceinline__ void async_copy_b128(void* lds_dst, const void* gsrc) {
  // low 32 bits of a generic LDS pointer are the wave-relative LDS offset
  unsigned lds_off = (unsigned)(unsigned long long)lds_dst;
  asm volatile("global_load_async_to_lds_b128 %0, %1, off"
               :: "v"(lds_off), "v"(gsrc) : "memory");
}
__device__ __forceinline__ void wait_async_le2() {
  asm volatile("s_wait_asynccnt 0x2" ::: "memory");
}
__device__ __forceinline__ void wait_async_0() {
  asm volatile("s_wait_asynccnt 0x0" ::: "memory");
}

// ---------------------------------------------------------------------------
// Shared GEMM core: 256 threads (8 waves), one 128x64 output tile per block.
// A: row-major [>=128][Kdim]; Bw: row-major [Kdim][>=64] (reduction = row idx).
// fp32 staged through registers (next chunk prefetched during WMMAs),
// converted to bf16 into double-buffered LDS; W tile stored transposed
// [col][k] so the B fragment is two contiguous 8-element reads per lane (k=8*half+0..7, 16+8*half+0..7).
// ---------------------------------------------------------------------------
__device__ __forceinline__ void gemm_core(const float* __restrict__ A, int lda,
                                          const float* __restrict__ Bw, int ldb,
                                          int Kdim, AccU acc[4]) {
  __shared__ __align__(16) __bf16 lA[2][128][32];  // 2 x 8 KB
  __shared__ __align__(16) __bf16 lB[2][64][32];   // 2 x 4 KB, transposed
  const int tid  = threadIdx.x;
  const int lane = tid & 31, wave = tid >> 5;
  const int half = lane >> 4, lq = lane & 15;
  const int arow = tid >> 3, ac4 = (tid & 7) << 2;   // A: 4 float4 / thread
  const int brow = tid >> 4, bc4 = (tid & 15) << 2;  // B: 2 float4 / thread

  float4 ra[4], rb[2];
  auto fetch = [&](int kt) {
#pragma unroll
    for (int i = 0; i < 4; ++i)
      ra[i] = *(const float4*)(A + (size_t)(arow + 32 * i) * lda + kt + ac4);
#pragma unroll
    for (int i = 0; i < 2; ++i)
      rb[i] = *(const float4*)(Bw + (size_t)(kt + brow + 16 * i) * ldb + bc4);
  };

  fetch(0);
  int buf = 0;
  for (int kt = 0; kt < Kdim; kt += 32) {
    // commit staged chunk (f32 -> bf16)
#pragma unroll
    for (int i = 0; i < 4; ++i) {
      lA[buf][arow + 32 * i][ac4 + 0] = (__bf16)ra[i].x;
      lA[buf][arow + 32 * i][ac4 + 1] = (__bf16)ra[i].y;
      lA[buf][arow + 32 * i][ac4 + 2] = (__bf16)ra[i].z;
      lA[buf][arow + 32 * i][ac4 + 3] = (__bf16)ra[i].w;
    }
#pragma unroll
    for (int i = 0; i < 2; ++i) {
      lB[buf][bc4 + 0][brow + 16 * i] = (__bf16)rb[i].x;
      lB[buf][bc4 + 1][brow + 16 * i] = (__bf16)rb[i].y;
      lB[buf][bc4 + 2][brow + 16 * i] = (__bf16)rb[i].z;
      lB[buf][bc4 + 3][brow + 16 * i] = (__bf16)rb[i].w;
    }
    __syncthreads();
    if (kt + 32 < Kdim) fetch(kt + 32);  // prefetch next chunk during WMMAs

    // A frag (16x32): lane row = lq; elems 0-7 -> k=8*half+0..7,
    //                 elems 8-15 -> k=16+8*half+0..7   (ISA 7.12.2)
    FragB a;
    const __bf16* ap = &lA[buf][wave * 16 + lq][half * 8];
    a.q[0] = *(const u32x4*)(ap);
    a.q[1] = *(const u32x4*)(ap + 16);
    // all 4 B frags first, then back-to-back WMMAs (one dscnt wait)
    FragB bf[4];
#pragma unroll
    for (int ct = 0; ct < 4; ++ct) {
      const __bf16* bp = &lB[buf][ct * 16 + lq][half * 8];
      bf[ct].q[0] = *(const u32x4*)(bp);
      bf[ct].q[1] = *(const u32x4*)(bp + 16);
    }
#pragma unroll
    for (int ct = 0; ct < 4; ++ct)
      acc[ct].v = wmma_bf16(a.v, bf[ct].v, acc[ct].v);

    buf ^= 1;
    __syncthreads();
  }
}

// ---------------------------------------------------------------------------
// Kernel 1: input projections.  qval/kval -> [bh][c][64] bf16,
// vval stored transposed -> [bh][64][C] bf16 (so PV B-frag reads contiguous).
// ---------------------------------------------------------------------------
__global__ __launch_bounds__(256) void proj_kernel(
    const float* __restrict__ qin, const float* __restrict__ kvin,
    const float* __restrict__ wq, const float* __restrict__ wk,
    const float* __restrict__ wv,
    __bf16* __restrict__ qval, __bf16* __restrict__ kval,
    __bf16* __restrict__ vvalT) {
  const int ctile = blockIdx.x;           // C/128 tiles
  const int bh    = blockIdx.y;           // b*H + h
  const int sel   = blockIdx.z;           // 0=q 1=k 2=v
  const int h = bh & (H_ - 1);
  const int b = bh >> 4;

  const float* Ain = (sel == 0 ? qin : kvin) +
                     ((size_t)b * C_ + (size_t)ctile * 128) * M_;
  const float* W = (sel == 0 ? wq : (sel == 1 ? wk : wv)) + (size_t)h * M_ * 64;

  AccU acc[4];
#pragma unroll
  for (int i = 0; i < 4; ++i) { v8f z = {0,0,0,0,0,0,0,0}; acc[i].v = z; }

  gemm_core(Ain, M_, W, 64, M_, acc);

  const int lane = threadIdx.x & 31, wave = threadIdx.x >> 5;
  const int half = lane >> 4, lq = lane & 15;
  __bf16* outp; size_t sR, sC;
  if (sel == 0)      { outp = qval  + (size_t)bh * C_ * 64; sR = 64; sC = 1;  }
  else if (sel == 1) { outp = kval  + (size_t)bh * C_ * 64; sR = 64; sC = 1;  }
  else               { outp = vvalT + (size_t)bh * 64 * C_; sR = 1;  sC = C_; }
#pragma unroll
  for (int ct = 0; ct < 4; ++ct)
#pragma unroll
    for (int j = 0; j < 8; ++j) {
      int r   = ctile * 128 + wave * 16 + j + half * 8;  // row (c)
      int col = ct * 16 + lq;                            // col (k/v)
      outp[(size_t)r * sR + (size_t)col * sC] = (__bf16)acc[ct].f[j];
    }
}

// ---------------------------------------------------------------------------
// Kernel 2: flash-style attention, faithful to the reference orientation:
//   S[c,d] = kval . qval (rows=c keys, cols=d queries); -100 mask for c>d
//   added BEFORE the 1/sqrt(K) scale; softmax over c (all keys);
//   pre[d,v] = sum_c P[c,d] * vval[c,v].
// K/V chunks are DMA'd to double-buffered LDS with
// global_load_async_to_lds_b128 (ASYNCcnt), shared by all 8 waves; chunk i+1's
// copy overlaps chunk i's WMMAs/softmax. In D-layout the exp'd probabilities
// repack IN-LANE into the A-fragment of P^T (no cross-lane transpose).
// ---------------------------------------------------------------------------
__global__ __launch_bounds__(256) void attn_kernel(
    const __bf16* __restrict__ qval, const __bf16* __restrict__ kval,
    const __bf16* __restrict__ vvalT, float* __restrict__ pre) {
  __shared__ __align__(16) __bf16 lK[2][32][64];  // 2 x 4 KB, [c][k]
  __shared__ __align__(16) __bf16 lV[2][64][32];  // 2 x 4 KB, [v][c]
  const int dtile = blockIdx.x;   // C/128
  const int bh    = blockIdx.y;
  const int h = bh & (H_ - 1);
  const int b = bh >> 4;
  const int tid  = threadIdx.x;
  const int lane = tid & 31, wave = tid >> 5;
  const int half = lane >> 4, lq = lane & 15;
  const int d0   = dtile * 128 + wave * 16;
  const int dabs = d0 + lq;       // this lane's query column in S tiles

  const __bf16* Q  = qval  + (size_t)bh * C_ * 64;
  const __bf16* Kp = kval  + (size_t)bh * C_ * 64;
  const __bf16* Vp = vvalT + (size_t)bh * 64 * C_;   // [v][c]

  // per-thread DMA slots: K chunk = 256 x 16B, V chunk = 256 x 16B
  const int kr = tid >> 3, ks = (tid & 7) << 3;   // K: row c, 8-elem segment
  const int vr = tid >> 2, vs = (tid & 3) << 3;   // V: row v, 8-elem segment
  auto issue_chunk = [&](int c0, int buf) {
    async_copy_b128(&lK[buf][kr][ks], Kp + (size_t)(c0 + kr) * 64 + ks);
    async_copy_b128(&lV[buf][vr][vs], Vp + (size_t)vr * C_ + c0 + vs);
  };

  // Q as B-operand of S = K x Q^T: col d = lq, elems i -> k = base+16*half+i
  FragB Qb[2];
#pragma unroll
  for (int kh = 0; kh < 2; ++kh) {
    const __bf16* p = Q + (size_t)(d0 + lq) * 64 + kh * 32 + half * 16;
    Qb[kh].q[0] = *(const u32x4*)p;
    Qb[kh].q[1] = *(const u32x4*)(p + 8);
  }

  AccU O[4];
#pragma unroll
  for (int i = 0; i < 4; ++i) { v8f z = {0,0,0,0,0,0,0,0}; O[i].v = z; }
  float mrun = -3.0e38f, srun = 0.f;
  const float scale = 0.125f;  // 1/sqrt(64)

  issue_chunk(0, 0);
  int buf = 0;
  for (int c0 = 0; c0 < C_; c0 += 32) {
    if (c0 + 32 < C_) {
      issue_chunk(c0 + 32, buf ^ 1);  // DMA next chunk behind this one's math
      wait_async_le2();               // oldest 2 (this chunk) complete
    } else {
      wait_async_0();
    }
    __syncthreads();

    // S tiles: keys c0..c0+15 (S[0]) and c0+16..c0+31 (S[1]); reduce k=64.
    AccU S[2];
#pragma unroll
    for (int i = 0; i < 2; ++i) { v8f z = {0,0,0,0,0,0,0,0}; S[i].v = z; }
#pragma unroll
    for (int ct = 0; ct < 2; ++ct)
#pragma unroll
      for (int kh = 0; kh < 2; ++kh) {
        FragB a;  // kval A-frag: row c = lq
        const __bf16* p = &lK[buf][ct * 16 + lq][kh * 32 + half * 16];
        a.q[0] = *(const u32x4*)p;
        a.q[1] = *(const u32x4*)(p + 8);
        S[ct].v = wmma_bf16(a.v, Qb[kh].v, S[ct].v);
      }
    // V B-frags early (hide LDS latency behind the softmax VALU work)
    FragB bv[4];
#pragma unroll
    for (int vt = 0; vt < 4; ++vt) {
      const __bf16* vp = &lV[buf][vt * 16 + lq][half * 8];
      bv[vt].q[0] = *(const u32x4*)vp;
      bv[vt].q[1] = *(const u32x4*)(vp + 16);
    }

    // mask + scale + online softmax over key axis c (per query column dabs)
    float vals[16];
    float cmax = -3.0e38f;
#pragma unroll
    for (int j = 0; j < 8; ++j) {
      int cr = c0 + j + half * 8;  // key row of S[0] element j
      float l0 = (S[0].f[j] + ((cr      > dabs) ? -100.f : 0.f)) * scale;
      float l1 = (S[1].f[j] + ((cr + 16 > dabs) ? -100.f : 0.f)) * scale;
      vals[j] = l0; vals[j + 8] = l1;
      cmax = fmaxf(cmax, fmaxf(l0, l1));
    }
    cmax = fmaxf(cmax, __shfl_xor(cmax, 16, 32));   // join row halves
    float mnew   = fmaxf(mrun, cmax);
    float fscale = __expf(mrun - mnew);
    float csum = 0.f;
    FragB pa;  // A-frag of P^T: elems 0-7 from S[0], 8-15 from S[1] (in-lane!)
#pragma unroll
    for (int i = 0; i < 16; ++i) {
      float p = __expf(vals[i] - mnew);
      csum += p;
      pa.e[i] = (__bf16)p;
    }
    csum += __shfl_xor(csum, 16, 32);
    srun = srun * fscale + csum;
    mrun = mnew;

    // rescale O (rows are d -> broadcast per-column stats across lanes)
#pragma unroll
    for (int j = 0; j < 8; ++j) {
      float fj = __shfl(fscale, j + 8 * half, 32);
#pragma unroll
      for (int vt = 0; vt < 4; ++vt) O[vt].f[j] *= fj;
    }

    // O += P^T x V
#pragma unroll
    for (int vt = 0; vt < 4; ++vt)
      O[vt].v = wmma_bf16(pa.v, bv[vt].v, O[vt].v);

    buf ^= 1;
    __syncthreads();   // safe to overwrite this buffer next iteration
  }

  // normalize and store pre as [b][d][h][v] fp32
#pragma unroll
  for (int j = 0; j < 8; ++j) {
    float sj  = __shfl(srun, j + 8 * half, 32);
    float inv = 1.0f / sj;
    int drow  = d0 + j + half * 8;
#pragma unroll
    for (int vt = 0; vt < 4; ++vt) {
      pre[(((size_t)b * C_ + drow) * H_ + h) * V_ + (vt * 16 + lq)] =
          O[vt].f[j] * inv;
    }
  }
}

// ---------------------------------------------------------------------------
// Kernel 3: out[b,c,m] = sum_{h,v} pre[b,c,h,v] * wo[h,v,m]
// = plain GEMM (4096 x 1024) x (1024 x 1024), fp32 output.
// ---------------------------------------------------------------------------
__global__ __launch_bounds__(256) void outproj_kernel(
    const float* __restrict__ pre, const float* __restrict__ wo,
    float* __restrict__ out) {
  const int rtile = blockIdx.x;   // (B*C)/128 = 32
  const int ctile = blockIdx.y;   // M/64 = 16
  const float* A = pre + (size_t)rtile * 128 * (H_ * V_);
  const float* W = wo + (size_t)ctile * 64;

  AccU acc[4];
#pragma unroll
  for (int i = 0; i < 4; ++i) { v8f z = {0,0,0,0,0,0,0,0}; acc[i].v = z; }

  gemm_core(A, H_ * V_, W, M_, H_ * V_, acc);

  const int lane = threadIdx.x & 31, wave = threadIdx.x >> 5;
  const int half = lane >> 4, lq = lane & 15;
#pragma unroll
  for (int ct = 0; ct < 4; ++ct)
#pragma unroll
    for (int j = 0; j < 8; ++j) {
      int r   = rtile * 128 + wave * 16 + j + half * 8;
      int col = ctile * 64 + ct * 16 + lq;
      out[(size_t)r * M_ + col] = acc[ct].f[j];
    }
}

// ---------------------------------------------------------------------------
extern "C" void kernel_launch(void* const* d_in, const int* in_sizes, int n_in,
                              void* d_out, int out_size, void* d_ws,
                              size_t ws_size, hipStream_t stream) {
  const float* kvin = (const float*)d_in[0];
  const float* qin  = (const float*)d_in[1];
  const float* wq   = (const float*)d_in[2];
  const float* wk   = (const float*)d_in[3];
  const float* wv   = (const float*)d_in[4];
  const float* wo   = (const float*)d_in[5];
  float* out = (float*)d_out;

  // workspace: qval(8MB) kval(8MB) vvalT(8MB) bf16 + pre(16MB) fp32 = 40MB
  const size_t nproj = (size_t)B_ * H_ * C_ * 64;
  __bf16* qval  = (__bf16*)d_ws;
  __bf16* kval  = qval + nproj;
  __bf16* vvalT = kval + nproj;
  float*  pre   = (float*)(vvalT + nproj);

  proj_kernel<<<dim3(C_ / 128, B_ * H_, 3), 256, 0, stream>>>(
      qin, kvin, wq, wk, wv, qval, kval, vvalT);
  attn_kernel<<<dim3(C_ / 128, B_ * H_), 256, 0, stream>>>(
      qval, kval, vvalT, pre);
  outproj_kernel<<<dim3((B_ * C_) / 128, M_ / 64), 256, 0, stream>>>(
      pre, wo, out);
}
